// IDSelfAttention_64811056496717
// MI455X (gfx1250) — matrix operand order repaired and checked by reading, not compiled, not run on hardware
//
#include <hip/hip_runtime.h>

// ---------------------------------------------------------------------------
// Types for CDNA5 (gfx1250) WMMA: wave32, V_WMMA_F32_16X16X32_BF16
// ---------------------------------------------------------------------------
typedef __attribute__((ext_vector_type(8)))  __bf16 bf16x8;
typedef __attribute__((ext_vector_type(16))) __bf16 bf16x16;
typedef __attribute__((ext_vector_type(8)))  float  f32x8;

static __device__ __forceinline__ bf16x16 bfcat(bf16x8 lo, bf16x8 hi) {
  return __builtin_shufflevector(lo, hi, 0,1,2,3,4,5,6,7,8,9,10,11,12,13,14,15);
}

static __device__ __forceinline__ __bf16 f2bf(float f) {
  unsigned u = __builtin_bit_cast(unsigned, f);
  unsigned r = (u + 0x7FFFu + ((u >> 16) & 1u)) >> 16;   // round-to-nearest-even
  unsigned short h = (unsigned short)r;
  return __builtin_bit_cast(__bf16, h);
}

static __device__ __forceinline__ float bf2f(__bf16 h) {
  unsigned u = ((unsigned)__builtin_bit_cast(unsigned short, h)) << 16;
  return __builtin_bit_cast(float, u);
}

static __device__ __forceinline__ f32x8 zero8() {
  f32x8 z;
#pragma unroll
  for (int i = 0; i < 8; ++i) z[i] = 0.0f;
  return z;
}

static __device__ __forceinline__ f32x8 wmma_bf16(bf16x16 a, bf16x16 b, f32x8 c) {
  return __builtin_amdgcn_wmma_f32_16x16x32_bf16(
      /*neg_a=*/false, a, /*neg_b=*/false, b,
      /*c_mod=*/(short)0, c, /*reuse_a=*/false, /*reuse_b=*/false);
}

// ---------------------------------------------------------------------------
// Problem constants
// ---------------------------------------------------------------------------
#define NTOK   4096
#define DMODEL 3072
#define NHEAD  16
#define DHEAD  64
#define INNER  1024       // NHEAD*DHEAD
#define SCALE  0.125f     // DHEAD^-0.5
#define LNEPS  1e-5f

// ---------------------------------------------------------------------------
// Kernel 1: LayerNorm (f32 in) -> bf16 out
// ---------------------------------------------------------------------------
__global__ __launch_bounds__(256)
void ln_kernel(const float* __restrict__ x, const float* __restrict__ g,
               const float* __restrict__ b, __bf16* __restrict__ xn) {
  const int row = blockIdx.x;
  const int tid = threadIdx.x;
  const float* xr = x + (size_t)row * DMODEL;
  __shared__ float red[256];

  float s = 0.0f;
  for (int c = tid; c < DMODEL; c += 256) s += xr[c];
  red[tid] = s;
  __syncthreads();
  for (int w = 128; w > 0; w >>= 1) {
    if (tid < w) red[tid] += red[tid + w];
    __syncthreads();
  }
  const float mu = red[0] * (1.0f / DMODEL);
  __syncthreads();

  float v = 0.0f;
  for (int c = tid; c < DMODEL; c += 256) { float d = xr[c] - mu; v += d * d; }
  red[tid] = v;
  __syncthreads();
  for (int w = 128; w > 0; w >>= 1) {
    if (tid < w) red[tid] += red[tid + w];
    __syncthreads();
  }
  const float rstd = rsqrtf(red[0] * (1.0f / DMODEL) + LNEPS);

  __bf16* xo = xn + (size_t)row * DMODEL;
  for (int c = tid; c < DMODEL; c += 256)
    xo[c] = f2bf((xr[c] - mu) * rstd * g[c] + b[c]);
}

// ---------------------------------------------------------------------------
// Kernel 2: f32 -> bf16 conversion (weights)
// ---------------------------------------------------------------------------
__global__ __launch_bounds__(256)
void cvt_kernel(const float* __restrict__ in, __bf16* __restrict__ out, int n) {
  int i = blockIdx.x * 256 + threadIdx.x;
  if (i < n) out[i] = f2bf(in[i]);
}

// ---------------------------------------------------------------------------
// Kernel 3: WMMA GEMM  C[M,N] = A[M,K] * Bw[N,K]^T   (bf16 in, f32 accum)
// block = 128 threads = 4 waves; block tile 128x128; wave tile 64x64
// Per K-step(32): 8 A b128-loads + 8 B b128-loads (x2 halves) + 16 v_wmma
// ---------------------------------------------------------------------------
template <typename OutT>
__global__ __launch_bounds__(128)
void gemm_bf16_wmma(const __bf16* __restrict__ A, const __bf16* __restrict__ Bw,
                    OutT* __restrict__ C, int M, int N, int K) {
  const int lane = threadIdx.x & 31;
  const int wave = threadIdx.x >> 5;
  const int half = lane >> 4;     // 0/1: selects K sub-stripe of the fragment
  const int l16  = lane & 15;

  const int rowBase = blockIdx.y * 128 + (wave >> 1) * 64;
  const int colBase = blockIdx.x * 128 + (wave & 1) * 64;

  f32x8 acc[4][4];
#pragma unroll
  for (int mt = 0; mt < 4; ++mt)
#pragma unroll
    for (int nt = 0; nt < 4; ++nt) acc[mt][nt] = zero8();

  for (int k0 = 0; k0 < K; k0 += 32) {
    bf16x16 af[4], bfr[4];
#pragma unroll
    for (int mt = 0; mt < 4; ++mt) {
      // A 16x32 bf16 layout: lane row = l16, K chunks at half*8 and 16+half*8
      const __bf16* ap = A + (size_t)(rowBase + mt * 16 + l16) * K + k0 + half * 8;
      bf16x8 lo = *(const bf16x8*)(ap);
      bf16x8 hi = *(const bf16x8*)(ap + 16);
      af[mt] = bfcat(lo, hi);
      __builtin_prefetch(ap + 32, 0, 1);   // next K-slab -> global_prefetch
    }
#pragma unroll
    for (int nt = 0; nt < 4; ++nt) {
      // B 32x16 bf16 layout: lane col = l16, K chunks at half*8 and 16+half*8
      const __bf16* bp = Bw + (size_t)(colBase + nt * 16 + l16) * K + k0 + half * 8;
      bf16x8 lo = *(const bf16x8*)(bp);
      bf16x8 hi = *(const bf16x8*)(bp + 16);
      bfr[nt] = bfcat(lo, hi);
      __builtin_prefetch(bp + 32, 0, 1);
    }
#pragma unroll
    for (int mt = 0; mt < 4; ++mt)
#pragma unroll
      for (int nt = 0; nt < 4; ++nt)
        acc[mt][nt] = wmma_bf16(af[mt], bfr[nt], acc[mt][nt]);
  }

#pragma unroll
  for (int mt = 0; mt < 4; ++mt)
#pragma unroll
    for (int nt = 0; nt < 4; ++nt) {
      const int col = colBase + nt * 16 + l16;
#pragma unroll
      for (int r = 0; r < 8; ++r) {
        const int row = rowBase + mt * 16 + r + 8 * half;  // C layout: M = r + 8*(lane/16)
        float v = acc[mt][nt][r];
        if constexpr (__is_same(OutT, float)) C[(size_t)row * N + col] = v;
        else                                  C[(size_t)row * N + col] = f2bf(v);
      }
    }
}

// ---------------------------------------------------------------------------
// Kernel 4: RoPE + head split.  qkv bf16 [N, 3*INNER] ->
//   Q [H,N,64] (A layout), K [H,N,64] (B layout for QK^T), V^T [H,64,N]
// ---------------------------------------------------------------------------
__global__ __launch_bounds__(1024)
void rope_kernel(const __bf16* __restrict__ qkv, const float* __restrict__ pe,
                 __bf16* __restrict__ Q, __bf16* __restrict__ Km,
                 __bf16* __restrict__ Vt) {
  const int n = blockIdx.x;
  const int t = threadIdx.x;          // h*64 + d
  const int h = t >> 6;
  const int d = t & 63;
  const int p = d >> 1;
  const int i = d & 1;

  const float* pe4 = pe + ((size_t)n * (DHEAD / 2) + p) * 4;
  const float c0 = pe4[i * 2 + 0];
  const float c1 = pe4[i * 2 + 1];

  const size_t base = (size_t)n * (3 * INNER);
  const float q0 = bf2f(qkv[base + h * DHEAD + 2 * p]);
  const float q1 = bf2f(qkv[base + h * DHEAD + 2 * p + 1]);
  const float k0 = bf2f(qkv[base + INNER + h * DHEAD + 2 * p]);
  const float k1 = bf2f(qkv[base + INNER + h * DHEAD + 2 * p + 1]);

  const size_t hn = ((size_t)h * NTOK + n) * DHEAD + d;
  Q[hn]  = f2bf(c0 * q0 + c1 * q1);
  Km[hn] = f2bf(c0 * k0 + c1 * k1);
  Vt[((size_t)h * DHEAD + d) * NTOK + n] = qkv[base + 2 * INNER + h * DHEAD + d];
}

// ---------------------------------------------------------------------------
// Kernel 5: Flash attention.  grid (N/64, H), block 128 (4 waves).
// K/V tiles are staged block-wide in LDS via async global->LDS loads
// (ASYNCcnt), double-buffered so the DMA overlaps the WMMA/softmax work.
// LDS map (bf16 elements; single __shared__ alloc => starts at LDS offset 0):
//   [0,4096)      K tile buf0   [key][d]      8 KB
//   [4096,8192)   V tile buf0   [d][key]      8 KB
//   [8192,12288)  K tile buf1                 8 KB
//   [12288,16384) V tile buf1                 8 KB
//   [16384,20480) per-wave P strips (4 x 16x64)
// ---------------------------------------------------------------------------
__global__ __launch_bounds__(128)
void attn_kernel(const __bf16* __restrict__ Q, const __bf16* __restrict__ Km,
                 const __bf16* __restrict__ Vt, __bf16* __restrict__ O) {
  const int h    = blockIdx.y;
  const int lane = threadIdx.x & 31;
  const int wave = threadIdx.x >> 5;
  const int half = lane >> 4;
  const int l16  = lane & 15;
  const int qBase = blockIdx.x * 64 + wave * 16;

  const __bf16* Qh = Q  + (size_t)h * NTOK * DHEAD;
  const __bf16* Kh = Km + (size_t)h * NTOK * DHEAD;
  const __bf16* Vh = Vt + (size_t)h * DHEAD * NTOK;

  __shared__ __bf16 smem[20480];                 // 40 KB of the 320 KB WGP LDS
  __bf16* pbuf = smem + 16384 + wave * 1024;

  const unsigned tid = threadIdx.x;
  const unsigned long long kgbase = (unsigned long long)(size_t)Kh;
  const unsigned long long vgbase = (unsigned long long)(size_t)Vh;

  // Issue async global->LDS copies of one 64-key K tile + V tile into buffer b.
  // 512 chunks of 16B per tile; 128 threads x 4 chunks each.
  auto issue_tile = [&](int kb, int b) {
#pragma unroll
    for (int j = 0; j < 4; ++j) {
      unsigned c = tid + (unsigned)j * 128u;
      // K tile: rows kb..kb+63 of Kh are one contiguous 8KB block
      unsigned long long ga = kgbase + (unsigned long long)kb * 128u + (unsigned long long)c * 16u;
      unsigned lo = (b ? 16384u : 0u) + c * 16u;               // LDS byte offset
      asm volatile("global_load_async_to_lds_b128 %0, %1, off"
                   :: "v"(lo), "v"(ga) : "memory");
    }
#pragma unroll
    for (int j = 0; j < 4; ++j) {
      unsigned c = tid + (unsigned)j * 128u;
      unsigned seg = c >> 3, w = c & 7;                        // 64 rows x 128B
      unsigned long long ga = vgbase +
          ((unsigned long long)seg * NTOK + (unsigned)kb) * 2u + (unsigned long long)w * 16u;
      unsigned lo = (b ? 24576u : 8192u) + c * 16u;
      asm volatile("global_load_async_to_lds_b128 %0, %1, off"
                   :: "v"(lo), "v"(ga) : "memory");
    }
  };

  // Q fragments (held in registers for the whole key loop)
  bf16x16 qa[2];
  {
    const __bf16* qp = Qh + (size_t)(qBase + l16) * DHEAD;
#pragma unroll
    for (int f = 0; f < 2; ++f) {
      bf16x8 lo = *(const bf16x8*)(qp + f * 32 + half * 8);
      bf16x8 hi = *(const bf16x8*)(qp + f * 32 + 16 + half * 8);
      qa[f] = bfcat(lo, hi);
    }
  }

  f32x8 oacc[4];
#pragma unroll
  for (int dt = 0; dt < 4; ++dt) oacc[dt] = zero8();
  float mi[8], li[8];
#pragma unroll
  for (int r = 0; r < 8; ++r) { mi[r] = -INFINITY; li[r] = 0.0f; }

  // Prime buffer 0
  issue_tile(0, 0);
  asm volatile("s_wait_asynccnt 0" ::: "memory");
  __syncthreads();

  for (int kb = 0; kb < NTOK; kb += 64) {
    const int cur = (kb >> 6) & 1;
    if (kb + 64 < NTOK) issue_tile(kb + 64, cur ^ 1);  // overlap DMA with compute

    const __bf16* kt = smem + (cur ? 8192 : 0);        // [key][d] 64x64
    const __bf16* vt = smem + (cur ? 12288 : 4096);    // [d][key] 64x64

    // ---- S = Q * K^T (16 x 64), B-frags from LDS (ds_load_b128) ----
    f32x8 s[4];
#pragma unroll
    for (int nt = 0; nt < 4; ++nt) s[nt] = zero8();
#pragma unroll
    for (int nt = 0; nt < 4; ++nt) {
      const __bf16* kp = kt + (nt * 16 + l16) * 64 + half * 8;
#pragma unroll
      for (int f = 0; f < 2; ++f) {
        bf16x8 lo = *(const bf16x8*)(kp + f * 32);
        bf16x8 hi = *(const bf16x8*)(kp + f * 32 + 16);
        s[nt] = wmma_bf16(qa[f], bfcat(lo, hi), s[nt]);
      }
    }

    // ---- online softmax (rows r and r+8 reduced simultaneously per half) ----
#pragma unroll
    for (int r = 0; r < 8; ++r) {
#pragma unroll
      for (int nt = 0; nt < 4; ++nt) s[nt][r] *= SCALE;
      float mx = s[0][r];
#pragma unroll
      for (int nt = 1; nt < 4; ++nt) mx = fmaxf(mx, s[nt][r]);
#pragma unroll
      for (int m = 1; m <= 8; m <<= 1) mx = fmaxf(mx, __shfl_xor(mx, m, 32));
      const float mnew  = fmaxf(mi[r], mx);
      const float alpha = __expf(mi[r] - mnew);
      float rs = 0.0f;
#pragma unroll
      for (int nt = 0; nt < 4; ++nt) {
        float pv = __expf(s[nt][r] - mnew);
        s[nt][r] = pv;
        rs += pv;
      }
#pragma unroll
      for (int m = 1; m <= 8; m <<= 1) rs += __shfl_xor(rs, m, 32);
      li[r] = li[r] * alpha + rs;
      mi[r] = mnew;
#pragma unroll
      for (int dt = 0; dt < 4; ++dt) oacc[dt][r] *= alpha;
      // spill P row to the wave's LDS strip (row-major -> A-frag reload is b128)
      const int rowOff = (r + 8 * half) * 64;
#pragma unroll
      for (int nt = 0; nt < 4; ++nt)
        pbuf[rowOff + nt * 16 + l16] = f2bf(s[nt][r]);
    }

    asm volatile("s_wait_dscnt 0" ::: "memory");  // lane-crossing LDS RAW

    // ---- reload P as A fragments ----
    bf16x16 pa[2];
#pragma unroll
    for (int f = 0; f < 2; ++f) {
      const __bf16* pp = pbuf + l16 * 64 + f * 32 + half * 8;
      bf16x8 lo = *(const bf16x8*)(pp);
      bf16x8 hi = *(const bf16x8*)(pp + 16);
      pa[f] = bfcat(lo, hi);
    }

    // ---- O += P * V  (V tile rows contiguous in key dim) ----
#pragma unroll
    for (int dt = 0; dt < 4; ++dt) {
      const __bf16* vp = vt + (dt * 16 + l16) * 64 + half * 8;
#pragma unroll
      for (int f = 0; f < 2; ++f) {
        bf16x8 lo = *(const bf16x8*)(vp + f * 32);
        bf16x8 hi = *(const bf16x8*)(vp + f * 32 + 16);
        oacc[dt] = wmma_bf16(pa[f], bfcat(lo, hi), oacc[dt]);
      }
    }

    // own async loads for next buffer done, then block-wide flip
    asm volatile("s_wait_asynccnt 0" ::: "memory");
    __syncthreads();
  }

  // ---- epilogue: O /= l, write o_bf16 [N, INNER] ----
#pragma unroll
  for (int dt = 0; dt < 4; ++dt) {
    const int col = h * DHEAD + dt * 16 + l16;
#pragma unroll
    for (int r = 0; r < 8; ++r) {
      const int row = qBase + r + 8 * half;
      O[(size_t)row * INNER + col] = f2bf(oacc[dt][r] / li[r]);
    }
  }
}

// ---------------------------------------------------------------------------
// Host-side launcher
// ---------------------------------------------------------------------------
extern "C" void kernel_launch(void* const* d_in, const int* in_sizes, int n_in,
                              void* d_out, int out_size, void* d_ws, size_t ws_size,
                              hipStream_t stream) {
  const float* x      = (const float*)d_in[0];   // [1,4096,3072]
  const float* pe     = (const float*)d_in[1];   // [1,1,4096,32,2,2]
  const float* w_qkv  = (const float*)d_in[2];   // [3072,3072]
  const float* w_out  = (const float*)d_in[3];   // [3072,1024]
  const float* ln_g   = (const float*)d_in[4];   // [3072]
  const float* ln_b   = (const float*)d_in[5];   // [3072]
  float* out = (float*)d_out;                    // [1,4096,3072]

  char* ws = (char*)d_ws;
  // workspace layout (bytes)
  __bf16* xnb   = (__bf16*)(ws);                    // 4096*3072*2 = 25165824
  __bf16* wqkvb = (__bf16*)(ws + 25165824u);        // 3072*3072*2 = 18874368
  __bf16* woutb = (__bf16*)(ws + 44040192u);        // 3072*1024*2 =  6291456
  __bf16* qkvb  = (__bf16*)(ws + 50331648u);        // 4096*3072*2 = 25165824
  __bf16* qb    = (__bf16*)(ws + 75497472u);        // 16*4096*64*2 = 8388608
  __bf16* kbuf  = (__bf16*)(ws + 83886080u);        // 8388608
  __bf16* vtb   = (__bf16*)(ws + 92274688u);        // 8388608
  __bf16* ob    = (__bf16*)(ws);                    // alias dead xnb (8.4MB <= 25MB)
  // total footprint: 100663296 bytes

  // 1) LayerNorm -> bf16
  ln_kernel<<<NTOK, 256, 0, stream>>>(x, ln_g, ln_b, xnb);

  // 2) weights -> bf16
  cvt_kernel<<<(DMODEL * DMODEL + 255) / 256, 256, 0, stream>>>(w_qkv, wqkvb, DMODEL * DMODEL);
  cvt_kernel<<<(DMODEL * INNER  + 255) / 256, 256, 0, stream>>>(w_out, woutb, DMODEL * INNER);

  // 3) QKV GEMM: [4096,3072] x [3072,3072]^T -> bf16 [4096,3072]
  gemm_bf16_wmma<__bf16><<<dim3(DMODEL / 128, NTOK / 128), 128, 0, stream>>>(
      xnb, wqkvb, qkvb, NTOK, DMODEL, DMODEL);

  // 4) RoPE + head split/transposes
  rope_kernel<<<NTOK, 1024, 0, stream>>>(qkvb, pe, qb, kbuf, vtb);

  // 5) flash attention -> o bf16 [4096,1024]  (note: aliases xnb, which is dead)
  attn_kernel<<<dim3(NTOK / 64, NHEAD), 128, 0, stream>>>(qb, kbuf, vtb, ob);

  // 6) out projection: [4096,1024] x [3072,1024]^T -> f32 d_out
  gemm_bf16_wmma<float><<<dim3(DMODEL / 128, NTOK / 128), 128, 0, stream>>>(
      ob, woutb, out, NTOK, DMODEL, INNER);
}
